// GlobalFilter_CF_dft_45775761441150
// MI455X (gfx1250) — matrix operand order repaired and checked by reading, not compiled, not run on hardware
//
#include <hip/hip_runtime.h>
#include <hip/hip_bf16.h>
#include <math.h>
#include <stdint.h>

typedef __attribute__((ext_vector_type(16))) _Float16 v16h;
typedef __attribute__((ext_vector_type(8)))  _Float16 v8h;
typedef __attribute__((ext_vector_type(2)))  _Float16 h2;
typedef __attribute__((ext_vector_type(8)))  float    v8f;

#define NTOK 3136          // 56*56 tokens
#define CCH  96            // channels
#define NFRQ 49            // rfft bins
#define K2   128           // padded interleaved (Yr,Yi) width = 2*64
#define B1_STRIDE 104      // 96 ch + 8 pad  (bank-conflict-free b128 reads)
#define B2_STRIDE 136      // 128 k + 8 pad
#define B1_ELEMS (128 * B1_STRIDE)   // 13312 halfs
#define B2_ELEMS (96  * B2_STRIDE)   // 13056 halfs
#define B1_CHUNKS 1664     // 16-byte chunks (26624 B)
#define B2_CHUNKS 1632     // 16-byte chunks (26112 B)
#define YB_STRIDE 136
#define ROWS_PER_BLOCK 128
#define NBLOCKS (128 * NTOK / ROWS_PER_BLOCK)   // 3136

// ---------------------------------------------------------------------------
// Init: write f16 DFT bases (padded layouts) into workspace.
// fwd (B1t): [n=0..127][c=0..103], B1t[n][c] = Fr/Fi[c][n]   (n<64 Re, >=64 Im)
// inv (B2t): [c=0..95 ][k=0..135], rows INTERLEAVED: k=2f -> Gr[f][c],
//                                                    k=2f+1 -> Gi[f][c]
// ---------------------------------------------------------------------------
__global__ void gf_init_basis(_Float16* __restrict__ ws) {
    int idx = blockIdx.x * blockDim.x + threadIdx.x;
    const float inv96 = 0.10206207261596575f;          // 1/sqrt(96)
    const float w0    = 6.283185307179586f / 96.0f;    // 2*pi/96
    if (idx < B1_ELEMS) {
        int n = idx / B1_STRIDE;          // freq-component column
        int c = idx % B1_STRIDE;          // channel
        float v = 0.f;
        int  f  = n & 63;
        bool im = n >= 64;
        if (c < CCH && f < NFRQ) {
            float th = w0 * (float)(f * c);
            v = (im ? -sinf(th) : cosf(th)) * inv96;
        }
        ws[idx] = (_Float16)v;
    } else if (idx < B1_ELEMS + B2_ELEMS) {
        int j = idx - B1_ELEMS;
        int c = j / B2_STRIDE;            // output channel
        int k = j % B2_STRIDE;            // interleaved freq row
        float v = 0.f;
        if (k < K2) {
            int  f  = k >> 1;             // 0..63
            bool im = k & 1;              // even: Gr, odd: Gi
            if (f < NFRQ) {
                float alpha = (f == 0 || f == 48) ? 1.f : 2.f;
                float th = w0 * (float)(f * c);
                v = alpha * (im ? -sinf(th) : cosf(th)) * inv96;
            }
        }
        ws[idx] = (_Float16)v;
    }
}

// ---------------------------------------------------------------------------
// Main fused kernel: GEMM1 (x * DFT) -> complex scale by w[token] -> GEMM2
// 256 threads = 8 waves; each wave owns one 16-row WMMA tile (128 rows/block).
// ---------------------------------------------------------------------------
__global__ __launch_bounds__(256) void gf_main(const float* __restrict__ x,
                                               const float* __restrict__ wc,
                                               float* __restrict__ out,
                                               const _Float16* __restrict__ basis) {
    __shared__ _Float16 sB1[B1_ELEMS];
    __shared__ _Float16 sB2[B2_ELEMS];
    __shared__ _Float16 sY[8][16 * YB_STRIDE];

    const int  tid     = threadIdx.x;
    const int  wave    = tid >> 5;
    const int  lane    = tid & 31;
    const int  lrow    = lane & 15;              // M for A-frag, N for B/C/D
    const bool hihalf  = lane >= 16;
    const int  rowBase = blockIdx.x * ROWS_PER_BLOCK + wave * 16;

    // prefetch this wave's x tile while the basis staging is in flight
    __builtin_prefetch(x + (size_t)(rowBase + lrow) * CCH, 0, 1);

    // ---- stage bases into LDS via async DMA (no VGPR round-trip) ----
    {
        const uint32_t ldsB1 = (uint32_t)(uintptr_t)&sB1[0];
        const uint32_t ldsB2 = (uint32_t)(uintptr_t)&sB2[0];
        const uint64_t g     = (uint64_t)(uintptr_t)basis;
        for (int i = tid; i < B1_CHUNKS; i += 256) {
            uint32_t l = ldsB1 + (uint32_t)i * 16u;
            uint64_t a = g + (uint64_t)i * 16u;
            asm volatile("global_load_async_to_lds_b128 %0, %1, off"
                         :: "v"(l), "v"(a) : "memory");
        }
        for (int i = tid; i < B2_CHUNKS; i += 256) {
            uint32_t l = ldsB2 + (uint32_t)i * 16u;
            uint64_t a = g + (uint64_t)(B1_CHUNKS + i) * 16u;
            asm volatile("global_load_async_to_lds_b128 %0, %1, off"
                         :: "v"(l), "v"(a) : "memory");
        }
        asm volatile("s_wait_asynccnt 0" ::: "memory");
    }
    __syncthreads();

    // ================= GEMM1: [16 x 96] * [96 x 128] ======================
    v8f acc1[8];
    #pragma unroll
    for (int i = 0; i < 8; ++i)
        #pragma unroll
        for (int j = 0; j < 8; ++j) acc1[i][j] = 0.f;

    // A fragments (3 k-steps of 32), ISA 16-bit A layout:
    //   lanes 0-15:  K = ks*32 + {0..7, 16..23};  lanes 16-31: +8
    v16h a1[3];
    {
        const int arow = rowBase + lrow;
        const float4* xp = (const float4*)(x + (size_t)arow * CCH);
        #pragma unroll
        for (int ks = 0; ks < 3; ++ks) {
            const int c0 = ks * 32 + (hihalf ? 8 : 0);
            float4 f0 = xp[c0 / 4];
            float4 f1 = xp[c0 / 4 + 1];
            float4 f2 = xp[(c0 + 16) / 4];
            float4 f3 = xp[(c0 + 16) / 4 + 1];
            a1[ks][0]  = (_Float16)f0.x; a1[ks][1]  = (_Float16)f0.y;
            a1[ks][2]  = (_Float16)f0.z; a1[ks][3]  = (_Float16)f0.w;
            a1[ks][4]  = (_Float16)f1.x; a1[ks][5]  = (_Float16)f1.y;
            a1[ks][6]  = (_Float16)f1.z; a1[ks][7]  = (_Float16)f1.w;
            a1[ks][8]  = (_Float16)f2.x; a1[ks][9]  = (_Float16)f2.y;
            a1[ks][10] = (_Float16)f2.z; a1[ks][11] = (_Float16)f2.w;
            a1[ks][12] = (_Float16)f3.x; a1[ks][13] = (_Float16)f3.y;
            a1[ks][14] = (_Float16)f3.z; a1[ks][15] = (_Float16)f3.w;
        }
    }

    #pragma unroll
    for (int ks = 0; ks < 3; ++ks) {
        const int kb = ks * 32 + (hihalf ? 8 : 0);    // B layout: K = {0..7, 16..23} (+8 hi half)
        #pragma unroll
        for (int nt = 0; nt < 8; ++nt) {
            const _Float16* bp = &sB1[(nt * 16 + lrow) * B1_STRIDE + kb];
            v8h blo = *(const v8h*)bp;
            v8h bhi = *(const v8h*)(bp + 16);
            v16h b = __builtin_shufflevector(blo, bhi,
                0, 1, 2, 3, 4, 5, 6, 7, 8, 9, 10, 11, 12, 13, 14, 15);
            acc1[nt] = __builtin_amdgcn_wmma_f32_16x16x32_f16(
                false, a1[ks], false, b, (short)0, acc1[nt], false, false);
        }
    }

    // ============ fused per-token complex scale -> LDS bounce =============
    // rowBase is wave-uniform: force token math onto SALU via readfirstlane.
    // Offsets are premultiplied by NFRQ on the scalar side; the only per-lane
    // op per v is one cndmask.
    const int rowBaseU = __builtin_amdgcn_readfirstlane(rowBase);
    const int tokB     = rowBaseU % NTOK;                 // SALU modulo (uniform)
    const int mofs     = hihalf ? 8 : 0;
    int tokOff[8];
    #pragma unroll
    for (int v = 0; v < 8; ++v) {
        int tlo = tokB + v;      if (tlo >= NTOK) tlo -= NTOK;   // uniform/SALU
        int thi = tokB + v + 8;  if (thi >= NTOK) thi -= NTOK;   // uniform/SALU
        int slo = tlo * NFRQ;                                    // SALU mul
        int shi = thi * NFRQ;                                    // SALU mul
        tokOff[v] = hihalf ? shi : slo;                          // 1 cndmask
    }

    const float2* w2    = (const float2*)wc;
    _Float16*     ybase = &sY[wave][0] + mofs * YB_STRIDE + 2 * lrow;
    const bool    f48v  = (lrow == 0);   // nt=3: only f==48 is a real bin

    #pragma unroll
    for (int v = 0; v < 8; ++v) {
        const float2* wp = w2 + tokOff[v] + lrow;     // one base per v
        _Float16*     yp = ybase + v * YB_STRIDE;     // immediate offsets below
        #pragma unroll
        for (int nt = 0; nt < 4; ++nt) {
            float wr, wi;
            if (nt < 3) {                             // f = nt*16+lrow <= 47: valid
                float2 w = wp[nt * 16];               // +128 B immediates
                wr = w.x; wi = w.y;
            } else {                                  // clamped load @f=48, select
                float2 w = w2[tokOff[v] + 48];
                wr = f48v ? w.x : 0.f;
                wi = f48v ? w.y : 0.f;
            }
            float xr = acc1[nt][v];
            float xi = acc1[nt + 4][v];
            h2 p;
            p.x = (_Float16)(xr * wr - xi * wi);
            p.y = (_Float16)(xr * wi + xi * wr);
            *(h2*)(yp + nt * 32) = p;                 // +64 B immediates
        }
    }
    asm volatile("s_wait_dscnt 0" ::: "memory");      // same-wave LDS RAW

    // ================= GEMM2: [16 x 128] * [128 x 96] =====================
    v8f acc2[6];
    #pragma unroll
    for (int i = 0; i < 6; ++i)
        #pragma unroll
        for (int j = 0; j < 8; ++j) acc2[i][j] = 0.f;

    #pragma unroll
    for (int ks = 0; ks < 4; ++ks) {
        const int c0 = ks * 32 + (hihalf ? 8 : 0);
        const _Float16* ap = &sY[wave][0] + lrow * YB_STRIDE;
        v8h g0 = *(const v8h*)(ap + c0);
        v8h g1 = *(const v8h*)(ap + c0 + 16);
        v16h a2 = __builtin_shufflevector(g0, g1,
            0, 1, 2, 3, 4, 5, 6, 7, 8, 9, 10, 11, 12, 13, 14, 15);
        const int kb = ks * 32 + (hihalf ? 8 : 0);
        #pragma unroll
        for (int nt = 0; nt < 6; ++nt) {
            const _Float16* bp = &sB2[(nt * 16 + lrow) * B2_STRIDE + kb];
            v8h blo = *(const v8h*)bp;
            v8h bhi = *(const v8h*)(bp + 16);
            v16h b = __builtin_shufflevector(blo, bhi,
                0, 1, 2, 3, 4, 5, 6, 7, 8, 9, 10, 11, 12, 13, 14, 15);
            acc2[nt] = __builtin_amdgcn_wmma_f32_16x16x32_f16(
                false, a2, false, b, (short)0, acc2[nt], false, false);
        }
    }

    // ========== store [16 x 96] f32: v-outer, nt via immediates ===========
    float* orow = out + (size_t)rowBase * CCH + (size_t)mofs * CCH + lrow;
    #pragma unroll
    for (int v = 0; v < 8; ++v) {
        float* op = orow + (size_t)v * CCH;           // one base per v
        #pragma unroll
        for (int nt = 0; nt < 6; ++nt)
            op[nt * 16] = acc2[nt][v];                // +64 B immediates
    }
}

// ---------------------------------------------------------------------------
extern "C" void kernel_launch(void* const* d_in, const int* in_sizes, int n_in,
                              void* d_out, int out_size, void* d_ws, size_t ws_size,
                              hipStream_t stream) {
    const float* x  = (const float*)d_in[0];   // [B, N, C] f32
    const float* wc = (const float*)d_in[1];   // [N, 49, 2] f32
    float* out = (float*)d_out;                // [B, N, C] f32
    _Float16* basis = (_Float16*)d_ws;         // 52736 B of f16 bases

    const int initTotal = B1_ELEMS + B2_ELEMS; // 26368
    gf_init_basis<<<(initTotal + 255) / 256, 256, 0, stream>>>(basis);
    gf_main<<<NBLOCKS, 256, 0, stream>>>(x, wc, out, basis);
}
